// GNNDis_23270132810341
// MI455X (gfx1250) — matrix operand order repaired and checked by reading, not compiled, not run on hardware
//
#include <hip/hip_runtime.h>
#include <hip/hip_bf16.h>
#include <math.h>

// ---------------------------------------------------------------------------
// GNNDis forward for MI455X (gfx1250, wave32).
//
// GATs + value head are O(N); the N^2 pairwise FFs are factorized
// (first layer: P_i + Q_j precompute) so the hot kernel is one dense
// streaming pass over the 18.9 MB output (HBM-write bound at 23.3 TB/s).
// Second-layer GEMMs (hidden @ W2) run on v_wmma_f32_16x16x32_f16 with
// 16-pair tiles. B fragments (constant weights) are pre-packed in the exact
// per-lane WMMA register layout (2x global_load_b128 per fragment); hidden
// A tiles are built with per-lane contiguous b128 LDS stores. Softplus uses
// the hardware TRANS path (v_exp_f32/v_log_f32) since results feed an f16
// WMMA operand anyway.
// ---------------------------------------------------------------------------

#define N_NODES 768
#define E_DIM 32
#define C_DIM 4
#define GAMMA 0.9f

typedef __attribute__((ext_vector_type(16))) _Float16 v16h;
typedef __attribute__((ext_vector_type(8)))  float    v8f;

__device__ __forceinline__ float neg_softplus(float v) {
  // -log(1+exp(v)), stable form, hardware transcendentals (TRANS class)
  return -(fmaxf(v, 0.0f) + __logf(1.0f + __expf(-fabsf(v))));
}

// ---- WMMA A-fragment builder (CDNA5 ISA layout) -----------------
// 16-bit A matrix 16x32 (MxK): lanes 0-15 row M=lane, K={0..7,16..23};
// lanes 16-31 row M=lane-16, K={8..15,24..31}; 2 f16 per VGPR.
// K-runs are contiguous -> compiler emits 2x ds_load_b128 per fragment.
__device__ __forceinline__ v16h load_A_16x32(const _Float16* tile, int ldm, int lane) {
  const int m  = lane & 15;
  const int kb = (lane < 16) ? 0 : 8;
  v16h a;
#pragma unroll
  for (int v = 0; v < 8; ++v) {
    const int k0 = (v < 4) ? (kb + 2 * v) : (kb + 16 + 2 * (v - 4));
    a[2 * v]     = tile[m * ldm + k0];
    a[2 * v + 1] = tile[m * ldm + k0 + 1];
  }
  return a;
}

// ---- K0: pre-pack B fragments (W2 padded to 16 cols) in WMMA lane layout ---
// B 32x16 (KxN) fragment: lane = column n (lanes 0-15: K {0..7,16..23} of chunk,
// lanes 16-31: K {8..15,24..31}), element e of the v16h = K (kb + e + (e<8?0:8)).
__global__ __launch_bounds__(160) void gnndis_k0(
    const float* __restrict__ Wu2, const float* __restrict__ We2,
    _Float16* __restrict__ BuP, _Float16* __restrict__ BeP) {
  const int t = threadIdx.x;
  const int lane = t & 31;
  const int n  = lane & 15;
  const int kb = (lane < 16) ? 0 : 8;
  if (t < 64) {            // util: 2 K-chunks (K = 64)
    const int kc = t >> 5;
#pragma unroll
    for (int e = 0; e < 16; ++e) {
      const int k = kc * 32 + kb + e + (e < 8 ? 0 : 8);
      BuP[(kc * 32 + lane) * 16 + e] =
          (n < C_DIM) ? (_Float16)Wu2[k * C_DIM + n] : (_Float16)0.f;
    }
  } else {                 // ext: 3 K-chunks (K = 72 padded to 96)
    const int kc = (t - 64) >> 5;
#pragma unroll
    for (int e = 0; e < 16; ++e) {
      const int k = kc * 32 + kb + e + (e < 8 ? 0 : 8);
      BeP[(kc * 32 + lane) * 16 + e] =
          (n < C_DIM && k < 72) ? (_Float16)We2[k * C_DIM + n] : (_Float16)0.f;
    }
  }
}

// ---- K1: h = x @ W (per GAT), s/d attention logits -------------------------
__global__ __launch_bounds__(32) void gnndis_k1(
    const float* __restrict__ x, const float* __restrict__ Wg,
    const float* __restrict__ Wl, const float* __restrict__ asg,
    const float* __restrict__ adg, const float* __restrict__ asl,
    const float* __restrict__ adl, float* __restrict__ hg,
    float* __restrict__ hl, float* __restrict__ sg, float* __restrict__ dg,
    float* __restrict__ sl, float* __restrict__ dl) {
  const int n = blockIdx.x;
  const int e = threadIdx.x;  // 32 lanes = E dims
  const float* xr = x + n * 16;
  float ag = 0.f, al = 0.f;
#pragma unroll
  for (int f = 0; f < 16; ++f) {
    ag += xr[f] * Wg[f * E_DIM + e];
    al += xr[f] * Wl[f * E_DIM + e];
  }
  hg[n * E_DIM + e] = ag;
  hl[n * E_DIM + e] = al;
  float psg = ag * asg[e], pdg = ag * adg[e];
  float psl = al * asl[e], pdl = al * adl[e];
#pragma unroll
  for (int o = 16; o > 0; o >>= 1) {
    psg += __shfl_xor(psg, o, 32);
    pdg += __shfl_xor(pdg, o, 32);
    psl += __shfl_xor(psl, o, 32);
    pdl += __shfl_xor(pdl, o, 32);
  }
  if (e == 0) { sg[n] = psg; dg[n] = pdg; sl[n] = psl; dl[n] = pdl; }
}

// ---- K2: masked softmax attention + aggregation, one wave per (row, gat) ---
__global__ __launch_bounds__(32) void gnndis_k2(
    const float* __restrict__ hg, const float* __restrict__ hl,
    const float* __restrict__ sg, const float* __restrict__ dg,
    const float* __restrict__ sl, const float* __restrict__ dl,
    const float* __restrict__ adj, float* __restrict__ yg,
    float* __restrict__ yl) {
  const int i = blockIdx.x;
  const int g = blockIdx.y;  // 0 = global GAT, 1 = local GAT
  const int lane = threadIdx.x;
  const float* h = g ? hl : hg;
  const float* d = g ? dl : dg;
  const float  si = g ? sl[i] : sg[i];
  float* y = g ? yl : yg;
  const float* arow = adj + (size_t)i * N_NODES;

  float m = -1e30f;
  for (int j = lane; j < N_NODES; j += 32) {
    float e = si + d[j];
    e = (e >= 0.f) ? e : 0.2f * e;  // leaky_relu(.,0.2)
    e = (arow[j] > 0.f) ? e : -1e9f;
    m = fmaxf(m, e);
  }
#pragma unroll
  for (int o = 16; o > 0; o >>= 1) m = fmaxf(m, __shfl_xor(m, o, 32));

  float Z = 0.f;
  for (int j = lane; j < N_NODES; j += 32) {
    if (arow[j] > 0.f) {
      float e = si + d[j];
      e = (e >= 0.f) ? e : 0.2f * e;
      Z += __expf(e - m);
    }
  }
#pragma unroll
  for (int o = 16; o > 0; o >>= 1) Z += __shfl_xor(Z, o, 32);

  // lane owns embedding dim `lane`
  float acc = 0.f;
  for (int j = 0; j < N_NODES; ++j) {
    if (arow[j] > 0.f) {
      float e = si + d[j];
      e = (e >= 0.f) ? e : 0.2f * e;
      acc += __expf(e - m) * h[j * E_DIM + lane];
    }
  }
  y[i * E_DIM + lane] = acc / Z;
}

// ---- K3: value head + pairwise-FF first-layer factorization (P, Q) ---------
__global__ __launch_bounds__(96) void gnndis_k3(
    const float* __restrict__ yg, const float* __restrict__ yl,
    const float* __restrict__ Wv1, const float* __restrict__ bv1,
    const float* __restrict__ Wv2, const float* __restrict__ bv2,
    const float* __restrict__ Wu1, const float* __restrict__ We1,
    float* __restrict__ val, float* __restrict__ Pg, float* __restrict__ Qg,
    float* __restrict__ Pl, float* __restrict__ Ql) {
  const int n = blockIdx.x;
  const int t = threadIdx.x;
  __shared__ float syg[E_DIM], syl[E_DIM], sh1[64];
  if (t < E_DIM) { syg[t] = yg[n * E_DIM + t]; syl[t] = yl[n * E_DIM + t]; }
  __syncthreads();
  if (t < 64) {
    float accv = bv1[t], accp = 0.f, accq = 0.f;
#pragma unroll 8
    for (int e = 0; e < E_DIM; ++e) {
      const float yv = syg[e];
      accv += yv * Wv1[e * 64 + t];
      accp += yv * Wu1[e * 64 + t];
      accq += yv * Wu1[(E_DIM + e) * 64 + t];
    }
    sh1[t] = neg_softplus(accv);
    Pg[n * 64 + t] = accp;
    Qg[n * 64 + t] = accq;
  }
  if (t < 72) {
    float accp = 0.f, accq = 0.f;
#pragma unroll 8
    for (int e = 0; e < E_DIM; ++e) {
      const float yv = syl[e];
      accp += yv * We1[e * 72 + t];
      accq += yv * We1[(E_DIM + e) * 72 + t];
    }
    Pl[n * 72 + t] = accp;
    Ql[n * 72 + t] = accq;
  }
  __syncthreads();
  if (t < C_DIM) {
    float acc = bv2[t];
#pragma unroll 8
    for (int h = 0; h < 64; ++h) acc += sh1[h] * Wv2[h * C_DIM + t];
    val[n * C_DIM + t] = neg_softplus(acc);
  }
}

// ---- K4: pairwise kernel, one wave per (b, i, 16-j tile), WMMA second layer
__global__ __launch_bounds__(32) void gnndis_k4(
    const float* __restrict__ adj, const float* __restrict__ pi,
    const float* __restrict__ Pg, const float* __restrict__ Qg,
    const float* __restrict__ Pl, const float* __restrict__ Ql,
    const float* __restrict__ val, const float* __restrict__ bu1,
    const float* __restrict__ bu2, const float* __restrict__ We1,
    const float* __restrict__ be1, const float* __restrict__ be2,
    const _Float16* __restrict__ BuP, const _Float16* __restrict__ BeP,
    float* __restrict__ out) {
  const int jt = blockIdx.x, i = blockIdx.y, b = blockIdx.z;
  const int j0 = jt * 16;
  const int lane = threadIdx.x;

  __shared__ float sPg[64];         // Pg_i + bu1 (bias folded)
  __shared__ float sPl[72];         // Pl_i + be1 (bias folded)
  __shared__ float sWpi[4 * 72];    // We1 rows 64..67 (pi projection)
  __shared__ __align__(32) _Float16 huA[16 * 64];   // util hidden (16 x 64)
  __shared__ __align__(32) _Float16 heA[16 * 96];   // ext hidden (16 x 72, pad 96)

  for (int t = lane; t < 64; t += 32) sPg[t] = Pg[i * 64 + t] + bu1[t];
  for (int t = lane; t < 72; t += 32) sPl[t] = Pl[i * 72 + t] + be1[t];
  for (int t = lane; t < 4 * 72; t += 32) sWpi[t] = We1[64 * 72 + t];
  __syncthreads();

  // util hidden: lane -> row r = lane&15, contiguous cols cbase..cbase+31,
  // packed to 2x v16h -> ds_store_b128s; Qg reads are per-lane b128 runs.
  {
    const int r = lane & 15;
    const int cbase = (lane < 16) ? 0 : 32;
    const int j = j0 + r;
    const float* qrow = Qg + j * 64 + cbase;
    const float* pg   = sPg + cbase;
    v16h t0, t1;
#pragma unroll
    for (int q = 0; q < 16; ++q)
      t0[q] = (_Float16)neg_softplus(pg[q] + qrow[q]);
#pragma unroll
    for (int q = 0; q < 16; ++q)
      t1[q] = (_Float16)neg_softplus(pg[16 + q] + qrow[16 + q]);
    *(v16h*)&huA[r * 64 + cbase]      = t0;
    *(v16h*)&huA[r * 64 + cbase + 16] = t1;
  }
  // ext hidden: lane -> row r, contiguous cols cbase..cbase+47 (pad >=72 to 0)
  {
    const int r = lane & 15;
    const int cbase = (lane < 16) ? 0 : 48;
    const int j = j0 + r;
    const float* pij = pi + (((size_t)b * N_NODES + i) * N_NODES + j) * C_DIM;
    const float p0 = pij[0], p1 = pij[1], p2 = pij[2], p3 = pij[3];
    auto hcalc = [&](int c) -> _Float16 {
      if (c >= 72) return (_Float16)0.f;
      float v = sPl[c] + Ql[j * 72 + c]
              + p0 * sWpi[c]       + p1 * sWpi[72 + c]
              + p2 * sWpi[144 + c] + p3 * sWpi[216 + c];
      return (_Float16)neg_softplus(v);
    };
    v16h t0, t1, t2;
#pragma unroll
    for (int q = 0; q < 16; ++q) t0[q] = hcalc(cbase + q);
#pragma unroll
    for (int q = 0; q < 16; ++q) t1[q] = hcalc(cbase + 16 + q);
#pragma unroll
    for (int q = 0; q < 16; ++q) t2[q] = hcalc(cbase + 32 + q);
    *(v16h*)&heA[r * 96 + cbase]      = t0;
    *(v16h*)&heA[r * 96 + cbase + 16] = t1;
    *(v16h*)&heA[r * 96 + cbase + 32] = t2;
  }
  __syncthreads();

  // Second layers: (16 x K) @ (K x 16) via v_wmma_f32_16x16x32_f16.
  // B fragments come pre-packed from L2: one v16h (2x global_load_b128)/chunk.
  const v16h* Bu = (const v16h*)BuP;
  const v16h* Be = (const v16h*)BeP;
  v8f cu = {};
#pragma unroll
  for (int kc = 0; kc < 2; ++kc) {  // K = 64
    v16h a = load_A_16x32(huA + kc * 32, 64, lane);
    v16h w = Bu[kc * 32 + lane];
    cu = __builtin_amdgcn_wmma_f32_16x16x32_f16(false, a, false, w,
                                                (short)0, cu, false, false);
  }
  v8f ce = {};
#pragma unroll
  for (int kc = 0; kc < 3; ++kc) {  // K = 96 (72 padded)
    v16h a = load_A_16x32(heA + kc * 32, 96, lane);
    v16h w = Be[kc * 32 + lane];
    ce = __builtin_amdgcn_wmma_f32_16x16x32_f16(false, a, false, w,
                                                (short)0, ce, false, false);
  }

  // D layout: lane col n = lane&15; VGPR r holds row m = (lane<16?0:8)+r
  const int n = lane & 15;
  if (n < C_DIM) {
    const int mbase = (lane < 16) ? 0 : 8;
    const float vi = val[i * C_DIM + n];
#pragma unroll
    for (int r = 0; r < 8; ++r) {
      const int j = j0 + mbase + r;
      const float u  = neg_softplus(cu[r] + bu2[n]);
      const float ex = neg_softplus(ce[r] + be2[n]);
      const float a  = adj[(size_t)i * N_NODES + j];
      const float o  = a * (u + ex) + GAMMA * val[j * C_DIM + n] - vi;
      out[(((size_t)b * C_DIM + n) * N_NODES + i) * N_NODES + j] = o;
    }
  }
}

extern "C" void kernel_launch(void* const* d_in, const int* in_sizes, int n_in,
                              void* d_out, int out_size, void* d_ws,
                              size_t ws_size, hipStream_t stream) {
  const float* x       = (const float*)d_in[0];
  const float* pi      = (const float*)d_in[1];
  // d_in[2] = bs (always 2; precompute is batch-invariant anyway)
  const float* adj     = (const float*)d_in[3];
  const float* Wg      = (const float*)d_in[4];
  const float* a_src_g = (const float*)d_in[5];
  const float* a_dst_g = (const float*)d_in[6];
  const float* Wl      = (const float*)d_in[7];
  const float* a_src_l = (const float*)d_in[8];
  const float* a_dst_l = (const float*)d_in[9];
  const float* Wu1     = (const float*)d_in[10];
  const float* bu1     = (const float*)d_in[11];
  const float* Wu2     = (const float*)d_in[12];
  const float* bu2     = (const float*)d_in[13];
  const float* We1     = (const float*)d_in[14];
  const float* be1     = (const float*)d_in[15];
  const float* We2     = (const float*)d_in[16];
  const float* be2     = (const float*)d_in[17];
  const float* Wv1     = (const float*)d_in[18];
  const float* bv1     = (const float*)d_in[19];
  const float* Wv2     = (const float*)d_in[20];
  const float* bv2     = (const float*)d_in[21];
  float* out = (float*)d_out;

  // Workspace layout. Packed B fragments first (32B-aligned region),
  // then float intermediates. Total ~1.26 MB.
  _Float16* BuP = (_Float16*)d_ws;             // 2*32*16 = 1024 f16 (2 KB)
  _Float16* BeP = BuP + 2 * 32 * 16;           // 3*32*16 = 1536 f16 (3 KB)
  float* ws  = (float*)(BeP + 3 * 32 * 16);
  float* hg  = ws;                 // 768*32
  float* hl  = hg + 768 * 32;      // 768*32
  float* sg  = hl + 768 * 32;      // 768
  float* dg  = sg + 768;           // 768
  float* sl  = dg + 768;           // 768
  float* dl  = sl + 768;           // 768
  float* yg  = dl + 768;           // 768*32
  float* yl  = yg + 768 * 32;      // 768*32
  float* val = yl + 768 * 32;      // 768*4
  float* Pg  = val + 768 * 4;      // 768*64
  float* Qg  = Pg + 768 * 64;      // 768*64
  float* Pl  = Qg + 768 * 64;      // 768*72
  float* Ql  = Pl + 768 * 72;      // 768*72

  gnndis_k0<<<dim3(1), dim3(160), 0, stream>>>(Wu2, We2, BuP, BeP);

  gnndis_k1<<<dim3(N_NODES), dim3(32), 0, stream>>>(
      x, Wg, Wl, a_src_g, a_dst_g, a_src_l, a_dst_l,
      hg, hl, sg, dg, sl, dl);

  gnndis_k2<<<dim3(N_NODES, 2), dim3(32), 0, stream>>>(
      hg, hl, sg, dg, sl, dl, adj, yg, yl);

  gnndis_k3<<<dim3(N_NODES), dim3(96), 0, stream>>>(
      yg, yl, Wv1, bv1, Wv2, bv2, Wu1, We1, val, Pg, Qg, Pl, Ql);

  gnndis_k4<<<dim3(N_NODES / 16, N_NODES, 2), dim3(32), 0, stream>>>(
      adj, pi, Pg, Qg, Pl, Ql, val, bu1, bu2, We1, be1, be2, BuP, BeP, out);
}
